// HGNN_Model_68298569941171
// MI455X (gfx1250) — matrix operand order repaired and checked by reading, not compiled, not run on hardware
//
#include <hip/hip_runtime.h>
#include <math.h>

// ---------------------------------------------------------------------------
// Problem constants (match reference)
// ---------------------------------------------------------------------------
#define N_C 50000
#define N_N 50000
#define NE  500000
#define D   128
#define ND  (50000 * 128)

// ---------------------------------------------------------------------------
// WMMA vector types (CDNA5 gfx1250, wave32)
// ---------------------------------------------------------------------------
typedef __attribute__((ext_vector_type(16))) __bf16 v16bf;
typedef __attribute__((ext_vector_type(8)))  float  v8f;

union BF16x16 {
    v16bf v;
    __bf16 h[16];
    uint4  q[2];
};

// ---------------------------------------------------------------------------
// zero fill (grid-stride, float4)
// ---------------------------------------------------------------------------
__global__ void zero_kernel(float4* __restrict__ p, int n4) {
    int i = blockIdx.x * blockDim.x + threadIdx.x;
    int stride = gridDim.x * blockDim.x;
    float4 z = make_float4(0.f, 0.f, 0.f, 0.f);
    for (; i < n4; i += stride) p[i] = z;
}

// ---------------------------------------------------------------------------
// weight pre-pack: Wb (column-major bf16)[j*128 + k] = bf16( W[k*128 + j] )
// Done once per call per weight matrix; Wb is tiny (32KB) and L2-resident,
// so the GEMM's B fragments become pure b128 vector loads with no conversion.
// ---------------------------------------------------------------------------
__global__ void pack_w_kernel(const float* __restrict__ W, __bf16* __restrict__ Wb) {
    int idx = blockIdx.x * 256 + threadIdx.x;   // 16384 elements
    int k = idx >> 7;          // row of W (contiguous read)
    int j = idx & 127;         // col of W
    Wb[(size_t)j * D + k] = (__bf16)W[(size_t)k * D + j];
}

// ---------------------------------------------------------------------------
// degree accumulation: one thread per edge
// ---------------------------------------------------------------------------
__global__ void degree_kernel(const int* __restrict__ src, const int* __restrict__ dst,
                              float* __restrict__ deg_out, float* __restrict__ deg_in, int E) {
    int e = blockIdx.x * blockDim.x + threadIdx.x;
    if (e >= E) return;
    if (deg_out) unsafeAtomicAdd(&deg_out[src[e]], 1.0f);
    unsafeAtomicAdd(&deg_in[dst[e]], 1.0f);
}

// ---------------------------------------------------------------------------
// edge scatter: one wave (32 lanes) per edge; each lane handles 4 floats.
// agg[dst] += x[src] * (deg_src ? rsqrt(max(deg_src[src],1)) : 1)
// ---------------------------------------------------------------------------
__global__ __launch_bounds__(256)
void scatter_kernel(const float* __restrict__ x, const int* __restrict__ src,
                    const int* __restrict__ dst, const float* __restrict__ deg_src,
                    float* __restrict__ agg, int E) {
    int edge = blockIdx.x * 8 + (threadIdx.x >> 5);
    if (edge >= E) return;
    int lane = threadIdx.x & 31;
    int s = src[edge];
    int d = dst[edge];
    float sc = 1.0f;
    if (deg_src) sc = rsqrtf(fmaxf(deg_src[s], 1.0f));
    const float4 v = ((const float4*)(x + (size_t)s * D))[lane];
    float* out = agg + (size_t)d * D + lane * 4;
    unsafeAtomicAdd(out + 0, v.x * sc);
    unsafeAtomicAdd(out + 1, v.y * sc);
    unsafeAtomicAdd(out + 2, v.z * sc);
    unsafeAtomicAdd(out + 3, v.w * sc);
}

// ---------------------------------------------------------------------------
// Fused WMMA GEMM:  Out[n,:] (+)= rowscale(n) * A[n,:] @ W  (+ bias) (relu?)
//   A: [nrows x 128] f32; Wb: [128 x 128] bf16 COLUMN-major (from pack_w);
//   Out: [nrows x 128] f32
//   scale_mode: 0=none, 1=rsqrt(max(deg,1)), 2=1/max(deg,1)
//   accumulate: Out += result (else Out = result); relu applied at final store
// One block = 256 threads = 8 waves; block handles 16 rows; wave w handles
// output columns [16w,16w+16). K=128 -> 4x v_wmma_f32_16x16x32_bf16 per wave.
// ---------------------------------------------------------------------------
__global__ __launch_bounds__(256)
void gemm128_wmma_kernel(const float* __restrict__ A, const float* __restrict__ deg,
                         int scale_mode, const __bf16* __restrict__ Wb,
                         const float* __restrict__ bias, float* __restrict__ Out,
                         int nrows, int accumulate, int relu) {
    const int lane = threadIdx.x & 31;
    const int wave = threadIdx.x >> 5;
    const int half = lane >> 4;    // which 16-lane half of the wave
    const int mloc = lane & 15;
    const int j0   = wave * 16;    // column tile base
    const int r0   = blockIdx.x * 16;
    const int n    = j0 + mloc;    // this lane's output column (B/C/D layout)

    // --- B fragments: 16-bit B matrix 32x16 per K-block of 32.
    //     packed element e<8 holds K = kb*32 + half*8 + e, e>=8 holds K = kb*32 + 16 + half*8 + (e-8), column = n.
    //     Wb is column-major bf16 -> two 16B runs (K and K+16) per fragment: 2x b128 loads.
    BF16x16 bf[4];
    const uint4* wp = (const uint4*)(Wb + (size_t)n * D);   // 16B-aligned
#pragma unroll
    for (int kb = 0; kb < 4; ++kb) {
        const int o8 = (kb * 32 + half * 8) >> 3;           // units of 8 bf16
        bf[kb].q[0] = wp[o8];
        bf[kb].q[1] = wp[o8 + 2];
    }

    // --- per-row scale for A
    const int m = r0 + mloc;             // A-fragment row for this lane
    const bool mvalid = (m < nrows);
    const int msafe = mvalid ? m : (nrows - 1);
    float rs = 1.0f;
    if (scale_mode == 1)      rs = rsqrtf(fmaxf(deg[msafe], 1.0f));
    else if (scale_mode == 2) rs = 1.0f / fmaxf(deg[msafe], 1.0f);
    if (!mvalid) rs = 0.0f;
    const float* Arow = A + (size_t)msafe * D;

    // --- A fragments: 16-bit A matrix 16x32 per K-block of 32.
    //     element e<8  holds K = kb*32 + half*8 + e
    //     element e>=8 holds K = kb*32 + 16 + half*8 + (e-8)
    BF16x16 af[4];
#pragma unroll
    for (int kb = 0; kb < 4; ++kb) {
        const int kA = kb * 32 + half * 8;
#pragma unroll
        for (int e = 0; e < 8; ++e) {
            af[kb].h[e]     = (__bf16)(Arow[kA + e] * rs);
            af[kb].h[e + 8] = (__bf16)(Arow[kA + 16 + e] * rs);
        }
    }

    // --- C init: bias (per column) + optional existing Out (accumulation).
    //     C/D 16x16 f32 layout: VGPR r: lanes 0-15 row = r, lanes 16-31 row = 8+r.
    const float binit = bias ? bias[n] : 0.0f;
    v8f c;
#pragma unroll
    for (int r = 0; r < 8; ++r) {
        const int row = r0 + half * 8 + r;
        float v = binit;
        if (accumulate && row < nrows) v += Out[(size_t)row * D + n];
        c[r] = v;
    }

    // --- 4 WMMA ops cover K = 128
#pragma unroll
    for (int kb = 0; kb < 4; ++kb)
        c = __builtin_amdgcn_wmma_f32_16x16x32_bf16(
                false, af[kb].v, false, bf[kb].v, (short)0, c, false, false);

    // --- store
#pragma unroll
    for (int r = 0; r < 8; ++r) {
        const int row = r0 + half * 8 + r;
        if (row < nrows) {
            float v = c[r];
            if (relu) v = fmaxf(v, 0.0f);
            Out[(size_t)row * D + n] = v;
        }
    }
}

// ---------------------------------------------------------------------------
// host-side helpers
// ---------------------------------------------------------------------------
static void launch_zero(float* p, int nfloats, hipStream_t s) {
    int n4 = nfloats / 4;
    int blocks = (n4 + 255) / 256;
    if (blocks > 4096) blocks = 4096;
    zero_kernel<<<blocks, 256, 0, s>>>((float4*)p, n4);
}

static void launch_gemm(const float* A, const float* deg, int scale_mode,
                        const __bf16* Wb, const float* bias, float* Out,
                        int nrows, int accumulate, int relu, hipStream_t s) {
    int blocks = (nrows + 15) / 16;   // 3125 for 50000 rows
    gemm128_wmma_kernel<<<blocks, 256, 0, s>>>(A, deg, scale_mode, Wb, bias, Out,
                                               nrows, accumulate, relu);
}

extern "C" void kernel_launch(void* const* d_in, const int* in_sizes, int n_in,
                              void* d_out, int out_size, void* d_ws, size_t ws_size,
                              hipStream_t stream) {
    const float* feat_C   = (const float*)d_in[0];
    const float* feat_N   = (const float*)d_in[1];
    const float* W1_cc    = (const float*)d_in[2];
    const float* b1_cc    = (const float*)d_in[3];
    const float* W1_cn    = (const float*)d_in[4];
    const float* b1_cn    = (const float*)d_in[5];
    const float* W1_self  = (const float*)d_in[6];
    const float* W1_neigh = (const float*)d_in[7];
    const float* b1_nn    = (const float*)d_in[8];
    const float* W2_cc    = (const float*)d_in[9];
    const float* b2_cc    = (const float*)d_in[10];
    const float* W2_cn    = (const float*)d_in[11];
    const float* b2_cn    = (const float*)d_in[12];
    const float* W2_self  = (const float*)d_in[13];
    const float* W2_neigh = (const float*)d_in[14];
    const float* b2_nn    = (const float*)d_in[15];
    const int*   cc_src   = (const int*)d_in[16];
    const int*   cc_dst   = (const int*)d_in[17];
    const int*   cn_src   = (const int*)d_in[18];
    const int*   cn_dst   = (const int*)d_in[19];
    const int*   nn_src   = (const int*)d_in[20];
    const int*   nn_dst   = (const int*)d_in[21];

    // workspace layout (floats)
    float* ws         = (float*)d_ws;
    float* deg_out_cc = ws;                 // [N_C]
    float* deg_in_cc  = ws + 50000;         // [N_C]
    float* deg_out_cn = ws + 100000;        // [N_C]
    float* deg_in_cn  = ws + 150000;        // [N_N]
    float* deg_in_nn  = ws + 200000;        // [N_N]
    float* agg        = ws + 250000;        // [50000 x 128] reusable
    float* hC         = agg + ND;           // [N_C x 128]
    float* hN         = hC + ND;            // [N_N x 128]
    __bf16* WbBase    = (__bf16*)(hN + ND); // 8 x [128 x 128] bf16 (col-major)

    __bf16* Wb1_cc    = WbBase + 0 * D * D;
    __bf16* Wb1_cn    = WbBase + 1 * D * D;
    __bf16* Wb1_self  = WbBase + 2 * D * D;
    __bf16* Wb1_neigh = WbBase + 3 * D * D;
    __bf16* Wb2_cc    = WbBase + 4 * D * D;
    __bf16* Wb2_cn    = WbBase + 5 * D * D;
    __bf16* Wb2_self  = WbBase + 6 * D * D;
    __bf16* Wb2_neigh = WbBase + 7 * D * D;

    float* oC = (float*)d_out;              // [N_C x 128]
    float* oN = oC + ND;                    // [N_N x 128]

    const int degBlocks  = (NE + 255) / 256;
    const int scatBlocks = (NE + 7) / 8;    // 8 edges (waves) per 256-thread block
    const int packBlocks = (D * D) / 256;   // 64

    // ---- pre-pack all weight matrices to col-major bf16 ----
    pack_w_kernel<<<packBlocks, 256, 0, stream>>>(W1_cc,    Wb1_cc);
    pack_w_kernel<<<packBlocks, 256, 0, stream>>>(W1_cn,    Wb1_cn);
    pack_w_kernel<<<packBlocks, 256, 0, stream>>>(W1_self,  Wb1_self);
    pack_w_kernel<<<packBlocks, 256, 0, stream>>>(W1_neigh, Wb1_neigh);
    pack_w_kernel<<<packBlocks, 256, 0, stream>>>(W2_cc,    Wb2_cc);
    pack_w_kernel<<<packBlocks, 256, 0, stream>>>(W2_cn,    Wb2_cn);
    pack_w_kernel<<<packBlocks, 256, 0, stream>>>(W2_self,  Wb2_self);
    pack_w_kernel<<<packBlocks, 256, 0, stream>>>(W2_neigh, Wb2_neigh);

    // ---- degrees (identical for both layers) ----
    launch_zero(ws, 250000, stream);
    degree_kernel<<<degBlocks, 256, 0, stream>>>(cc_src, cc_dst, deg_out_cc, deg_in_cc, NE);
    degree_kernel<<<degBlocks, 256, 0, stream>>>(cn_src, cn_dst, deg_out_cn, deg_in_cn, NE);
    degree_kernel<<<degBlocks, 256, 0, stream>>>(nn_src, nn_dst, nullptr,    deg_in_nn, NE);

    // ================= layer 1 =================
    // hC = relu( (sum_cc norm_src*feat_C) * norm_dst @ W1_cc + b1_cc )
    launch_zero(agg, ND, stream);
    scatter_kernel<<<scatBlocks, 256, 0, stream>>>(feat_C, cc_src, cc_dst, deg_out_cc, agg, NE);
    launch_gemm(agg, deg_in_cc, 1, Wb1_cc, b1_cc, hC, N_C, /*acc=*/0, /*relu=*/1, stream);

    // hN = relu( graphconv_cn(feat_C) + feat_N@W1_self + mean_nn(feat_N)@W1_neigh + b1_cn + b1_nn )
    launch_zero(agg, ND, stream);
    scatter_kernel<<<scatBlocks, 256, 0, stream>>>(feat_C, cn_src, cn_dst, deg_out_cn, agg, NE);
    launch_gemm(agg, deg_in_cn, 1, Wb1_cn, b1_cn, hN, N_N, 0, 0, stream);
    launch_gemm(feat_N, nullptr, 0, Wb1_self, b1_nn, hN, N_N, 1, 0, stream);
    launch_zero(agg, ND, stream);
    scatter_kernel<<<scatBlocks, 256, 0, stream>>>(feat_N, nn_src, nn_dst, nullptr, agg, NE);
    launch_gemm(agg, deg_in_nn, 2, Wb1_neigh, nullptr, hN, N_N, 1, /*relu=*/1, stream);

    // ================= layer 2 =================
    // oC = graphconv_cc(hC) @ W2_cc + b2_cc
    launch_zero(agg, ND, stream);
    scatter_kernel<<<scatBlocks, 256, 0, stream>>>(hC, cc_src, cc_dst, deg_out_cc, agg, NE);
    launch_gemm(agg, deg_in_cc, 1, Wb2_cc, b2_cc, oC, N_C, 0, 0, stream);

    // oN = graphconv_cn(hC) + hN@W2_self + mean_nn(hN)@W2_neigh + b2_cn + b2_nn
    launch_zero(agg, ND, stream);
    scatter_kernel<<<scatBlocks, 256, 0, stream>>>(hC, cn_src, cn_dst, deg_out_cn, agg, NE);
    launch_gemm(agg, deg_in_cn, 1, Wb2_cn, b2_cn, oN, N_N, 0, 0, stream);
    launch_gemm(hN, nullptr, 0, Wb2_self, b2_nn, oN, N_N, 1, 0, stream);
    launch_zero(agg, ND, stream);
    scatter_kernel<<<scatBlocks, 256, 0, stream>>>(hN, nn_src, nn_dst, nullptr, agg, NE);
    launch_gemm(agg, deg_in_nn, 2, Wb2_neigh, nullptr, oN, N_N, 1, 0, stream);
}
